// Bidirectional_LSTM_Encoder_19078244729406
// MI455X (gfx1250) — matrix operand order repaired and checked by reading, not compiled, not run on hardware
//
#include <hip/hip_runtime.h>
#include <hip/hip_bf16.h>
#include <cstdint>

#define DEVI __device__ __forceinline__

typedef __attribute__((ext_vector_type(16))) __bf16        v16bf;
typedef __attribute__((ext_vector_type(8)))  float          v8f;
typedef __attribute__((ext_vector_type(2)))  unsigned short u16x2;
typedef __attribute__((ext_vector_type(8)))  unsigned short u16x8;
typedef __attribute__((ext_vector_type(16))) unsigned short u16x16;

// CDNA5 async global->LDS path (ASYNCcnt-tracked), guarded so either toolchain compiles.
#if defined(__has_builtin)
#if __has_builtin(__builtin_amdgcn_global_load_async_to_lds_b32)
#define ASYNC_LDS 1
#endif
#if __has_builtin(__builtin_amdgcn_s_wait_asynccnt)
#define HAVE_WAIT_ASYNC_BUILTIN 1
#endif
#endif

typedef __attribute__((address_space(1))) int* gint_p;   // global int*
typedef __attribute__((address_space(3))) int* lint_p;   // LDS int*

static constexpr int    Bsz  = 64;
static constexpr int    Tlen = 512;
static constexpr int    Hdim = 1024;
static constexpr int    Lnum = 2;
static constexpr int    LDSP = 40;                                    // padded LDS row stride (elems)
static constexpr size_t HH     = (size_t)Hdim * Hdim;                 // gate stride in W (elems)
static constexpr size_t OUT_C  = (size_t)Bsz * Tlen * 2 * Hdim;       // c_out base in d_out
static constexpr size_t OUT_HL = 2 * OUT_C;                           // h_last base
static constexpr size_t OUT_CL = OUT_HL + (size_t)Bsz * 2 * Hdim;     // c_last base

DEVI unsigned short f2bf(float f) {
  unsigned u = __builtin_bit_cast(unsigned, f);
  unsigned r = u + 0x7FFFu + ((u >> 16) & 1u);     // RNE truncation
  return (unsigned short)(r >> 16);
}
DEVI float sigf(float x)     { return 1.0f / (1.0f + __expf(-x)); }
DEVI float tanhfast(float x) { float e = __expf(2.0f * x); return (e - 1.0f) / (e + 1.0f); }
DEVI v16bf ld_bf16x16(const unsigned short* p) {
  return __builtin_bit_cast(v16bf, *(const u16x16*)p);
}
DEVI void copy4_to_lds(unsigned short* ldsdst, const unsigned short* src) {
#if defined(ASYNC_LDS)
  __builtin_amdgcn_global_load_async_to_lds_b32(
      (gint_p)(uintptr_t)src,
      (lint_p)(uintptr_t)ldsdst,
      /*offset=*/0, /*cpol=*/0);
#else
  *(u16x2*)ldsdst = *(const u16x2*)src;
#endif
}
DEVI void wait_async_fill() {
#if defined(ASYNC_LDS)
#if defined(HAVE_WAIT_ASYNC_BUILTIN)
  __builtin_amdgcn_s_wait_asynccnt(0);
#else
  asm volatile("s_wait_asynccnt 0x0" ::: "memory");
#endif
#endif
}

// ---------------- prep kernels ----------------
__global__ void cvt_bf16_kernel(const float* __restrict__ s, unsigned short* __restrict__ d, size_t n) {
  size_t i = (size_t)blockIdx.x * blockDim.x + threadIdx.x;
  size_t st = (size_t)gridDim.x * blockDim.x;
  for (; i < n; i += st) d[i] = f2bf(s[i]);
}
__global__ void bias_sum_kernel(const float* __restrict__ a, const float* __restrict__ b,
                                float* __restrict__ d, int n) {
  int i = blockIdx.x * blockDim.x + threadIdx.x;
  if (i < n) d[i] = a[i] + b[i];
}
__global__ void zero_kernel(unsigned int* __restrict__ p, size_t nwords) {
  size_t i = (size_t)blockIdx.x * blockDim.x + threadIdx.x;
  size_t st = (size_t)gridDim.x * blockDim.x;
  for (; i < nwords; i += st) p[i] = 0u;
}

// ---------------- LSTM step: one (t, layer), both directions ----------------
// grid = 64 blocks x 256 threads (8 wave32). tile = blockIdx*8 + wave:
//   dir = blockIdx>>5 (uniform/block), mtile = (blockIdx>>3)&3 (uniform/block),
//   ntile = (blockIdx&7)*8 + wave.
__global__ void __launch_bounds__(256, 1)
lstm_step_kernel(const unsigned short* __restrict__ xbf,
                 const unsigned short* __restrict__ WihF, const unsigned short* __restrict__ WhhF,
                 const unsigned short* __restrict__ WihB, const unsigned short* __restrict__ WhhB,
                 const float* __restrict__ biasF, const float* __restrict__ biasB,
                 unsigned short* __restrict__ hstate, float* __restrict__ cstate,
                 unsigned short* __restrict__ houtF, unsigned short* __restrict__ houtB,
                 float* __restrict__ out, int t, int layer) {
  __shared__ unsigned short As[2][16 * LDSP];   // double-buffered 16 x 32 bf16 A tile, padded rows

  const int tid  = threadIdx.x;
  const int lane = tid & 31;
  const int wave = tid >> 5;
  const int dir  = blockIdx.x >> 5;                       // 0 = fwd, 1 = bwd (uniform per block)
  const int m0   = (((blockIdx.x >> 3) & 3)) << 4;        // batch tile base   (uniform per block)
  const int n0   = (((blockIdx.x & 7) << 3) + wave) << 4; // hidden tile base  (uniform per wave)
  const int pp   = t & 1;

  const unsigned short* WihL = (dir ? WihB : WihF) + (size_t)layer * 4 * HH;
  const unsigned short* WhhL = (dir ? WhhB : WhhF) + (size_t)layer * 4 * HH;
  const float*          bias = dir ? biasB : biasF;

  // K-chunk [0,1024): layer input (maybe feature-reversed); [1024,2048): recurrent state
  const unsigned short* Asrc0;
  size_t astride0;
  bool rev0 = false;
  if (layer == 0) {
    Asrc0 = xbf + (size_t)t * Hdim;          // x[b, t, k] at b*T*H + t*H + k
    astride0 = (size_t)Tlen * Hdim;
    rev0 = (dir == 1);                       // x_t[:, ::-1] feature reversal
  } else {
    Asrc0 = dir ? houtB : houtF;             // previous layer output, this t
    astride0 = Hdim;
  }
  const unsigned short* Asrc1 = hstate + ((size_t)layer * 2 + pp) * (size_t)Bsz * Hdim;

  // cooperative A-tile fill: 512 elems / 256 threads = 2 each (one b32 copy)
  const int fidx = tid * 2;
  const int fr   = fidx >> 5;        // row 0..15
  const int fcc  = fidx & 31;        // col 0..30 (even)
  auto fill_tile = [&](int buf, int k) {
    unsigned short* dst = &As[buf][fr * LDSP + fcc];
    if (k < Hdim) {
      const unsigned short* src = Asrc0 + (size_t)(m0 + fr) * astride0;
      if (rev0) {                                     // block-uniform branch
        dst[0] = src[Hdim - 1 - (k + fcc + 0)];
        dst[1] = src[Hdim - 1 - (k + fcc + 1)];
      } else {
        copy4_to_lds(dst, src + k + fcc);
      }
    } else {
      copy4_to_lds(dst, Asrc1 + (size_t)(m0 + fr) * Hdim + (k - Hdim) + fcc);
    }
  };

  v8f accI = {}, accF = {}, accG = {}, accO = {};

  const int lr    = lane & 15;
  const int ko    = (lane >> 4) << 4;        // A-frag K half (+0 / +16)
  const int kwsel = (lane >> 4) << 4;        // B-frag K half (+0 / +16)

  constexpr int NIT = (2 * Hdim) / 32;       // 64 K-chunks over input + recurrent GEMMs
  int buf = 0;
  fill_tile(0, 0);                           // prologue fill
  for (int it = 0; it < NIT; ++it) {
    const int kb = it * 32;
    wait_async_fill();                       // own async fills done (s_wait_asynccnt 0)
    __syncthreads();                         // everyone's fills visible
    if (it + 1 < NIT) fill_tile(buf ^ 1, kb + 32);   // overlap next fill with WMMAs

    // A fragment (16x32 bf16, contiguous k = 16*(lane/16) + s) from LDS
    u16x8 alo = *(const u16x8*)&As[buf][lr * LDSP + ko];
    u16x8 ahi = *(const u16x8*)&As[buf][lr * LDSP + 8 + ko];
    u16x16 av;
#pragma unroll
    for (int i = 0; i < 8; ++i) { av[i] = alo[i]; av[i + 8] = ahi[i]; }
    v16bf a = __builtin_bit_cast(v16bf, av);

    // B fragments: row n of W (g = x @ W^T), 16 contiguous bf16 per lane; 4 gates
    const unsigned short* Wp = (kb < Hdim) ? WihL : WhhL;
    const int kloc = kb & (Hdim - 1);
    const unsigned short* wrow = Wp + (size_t)(n0 + lr) * Hdim + (kloc + kwsel);
    __builtin_prefetch(wrow + 32, 0, 0);     // next K chunk -> global_prefetch_b8
    v16bf bI = ld_bf16x16(wrow + 0 * HH);
    v16bf bF = ld_bf16x16(wrow + 1 * HH);
    v16bf bG = ld_bf16x16(wrow + 2 * HH);
    v16bf bO = ld_bf16x16(wrow + 3 * HH);

    accI = __builtin_amdgcn_wmma_f32_16x16x32_bf16(false, a, false, bI, (short)0, accI, false, false);
    accF = __builtin_amdgcn_wmma_f32_16x16x32_bf16(false, a, false, bF, (short)0, accF, false, false);
    accG = __builtin_amdgcn_wmma_f32_16x16x32_bf16(false, a, false, bG, (short)0, accG, false, false);
    accO = __builtin_amdgcn_wmma_f32_16x16x32_bf16(false, a, false, bO, (short)0, accO, false, false);
    buf ^= 1;
  }

  // -------- pointwise LSTM epilogue, fully in-register --------
  const float* biasL = bias + layer * 4 * Hdim;
  const int nl = n0 + lr;
  const float bI = biasL[0 * Hdim + nl];
  const float bF = biasL[1 * Hdim + nl];
  const float bG = biasL[2 * Hdim + nl];
  const float bO = biasL[3 * Hdim + nl];

  const float*    cOld  = cstate + ((size_t)layer * 2 + pp) * (size_t)Bsz * Hdim;
  float*          cNew  = cstate + ((size_t)layer * 2 + (pp ^ 1)) * (size_t)Bsz * Hdim;
  unsigned short* hNew  = hstate + ((size_t)layer * 2 + (pp ^ 1)) * (size_t)Bsz * Hdim;
  unsigned short* hOutD = dir ? houtB : houtF;

#pragma unroll
  for (int r = 0; r < 8; ++r) {
    const int m = m0 + r + ((lane >> 4) << 3);     // C/D layout: VGPR r -> M=r (+8 for hi lanes)
    const size_t idx = (size_t)m * Hdim + nl;
    float iv = sigf(accI[r] + bI);
    float fv = sigf(accF[r] + bF);
    float gv = tanhfast(accG[r] + bG);
    float ov = sigf(accO[r] + bO);
    float cn = fv * cOld[idx] + iv * gv;
    float hn = ov * tanhfast(cn);
    unsigned short hb = f2bf(hn);
    hOutD[idx] = hb;                               // next layer's input
    if (dir) { hNew[idx] = hb; cNew[idx] = cn; }   // only backward cell carries state
    if (layer == Lnum - 1) {
      size_t o = (size_t)m * Tlen * 2 * Hdim + (size_t)t * 2 * Hdim + (size_t)dir * Hdim + nl;
      out[o] = hn;
      out[OUT_C + o] = cn;
      if (t == Tlen - 1) {
        size_t ol = (size_t)m * 2 * Hdim + (size_t)dir * Hdim + nl;
        out[OUT_HL + ol] = hn;
        out[OUT_CL + ol] = cn;
      }
    }
  }
}

// ---------------- host ----------------
extern "C" void kernel_launch(void* const* d_in, const int* in_sizes, int n_in,
                              void* d_out, int out_size, void* d_ws, size_t ws_size,
                              hipStream_t stream) {
  (void)in_sizes; (void)n_in; (void)out_size; (void)ws_size;
  const float* x     = (const float*)d_in[0];
  const float* Wih_f = (const float*)d_in[1];
  const float* Whh_f = (const float*)d_in[2];
  const float* bih_f = (const float*)d_in[3];
  const float* bhh_f = (const float*)d_in[4];
  const float* Wih_b = (const float*)d_in[5];
  const float* Whh_b = (const float*)d_in[6];
  const float* bih_b = (const float*)d_in[7];
  const float* bhh_b = (const float*)d_in[8];

  const size_t WE = (size_t)Lnum * 4 * Hdim * Hdim;   // 8,388,608 elems per weight tensor
  char* p = (char*)d_ws;
  unsigned short* wihf = (unsigned short*)p;            p += WE * 2;
  unsigned short* whhf = (unsigned short*)p;            p += WE * 2;
  unsigned short* wihb = (unsigned short*)p;            p += WE * 2;
  unsigned short* whhb = (unsigned short*)p;            p += WE * 2;
  float*          biasf = (float*)p;                    p += (size_t)Lnum * 4 * Hdim * 4;
  float*          biasb = (float*)p;                    p += (size_t)Lnum * 4 * Hdim * 4;
  unsigned short* xbf   = (unsigned short*)p;           p += (size_t)Bsz * Tlen * Hdim * 2;
  unsigned short* hstate = (unsigned short*)p;          p += (size_t)Lnum * 2 * Bsz * Hdim * 2;
  float*          cstate = (float*)p;                   p += (size_t)Lnum * 2 * Bsz * Hdim * 4;
  unsigned short* houtf  = (unsigned short*)p;          p += (size_t)Bsz * Hdim * 2;
  unsigned short* houtb  = (unsigned short*)p;          p += (size_t)Bsz * Hdim * 2;

  // weights + x -> bf16 (once per launch; deterministic)
  cvt_bf16_kernel<<<4096, 256, 0, stream>>>(Wih_f, wihf, WE);
  cvt_bf16_kernel<<<4096, 256, 0, stream>>>(Whh_f, whhf, WE);
  cvt_bf16_kernel<<<4096, 256, 0, stream>>>(Wih_b, wihb, WE);
  cvt_bf16_kernel<<<4096, 256, 0, stream>>>(Whh_b, whhb, WE);
  bias_sum_kernel<<<32, 256, 0, stream>>>(bih_f, bhh_f, biasf, Lnum * 4 * Hdim);
  bias_sum_kernel<<<32, 256, 0, stream>>>(bih_b, bhh_b, biasb, Lnum * 4 * Hdim);
  cvt_bf16_kernel<<<4096, 256, 0, stream>>>(x, xbf, (size_t)Bsz * Tlen * Hdim);

  // zero h/c state + layer-output buffers (contiguous region after xbf)
  const size_t zbytes = (size_t)Lnum * 2 * Bsz * Hdim * 2 + (size_t)Lnum * 2 * Bsz * Hdim * 4
                      + 2 * (size_t)Bsz * Hdim * 2;
  zero_kernel<<<1024, 256, 0, stream>>>((unsigned int*)hstate, zbytes / 4);

  for (int t = 0; t < Tlen; ++t)
    for (int layer = 0; layer < Lnum; ++layer)
      lstm_step_kernel<<<dim3(64), dim3(256), 0, stream>>>(
          xbf, wihf, whhf, wihb, whhb, biasf, biasb,
          hstate, cstate, houtf, houtb, (float*)d_out, t, layer);
}
